// SimplifiedMamba2Block_V2_1279900254548
// MI455X (gfx1250) — matrix operand order repaired and checked by reading, not compiled, not run on hardware
//
#include <hip/hip_runtime.h>
#include <hip/hip_bf16.h>

// ---------------------------------------------------------------------------
// Simplified Mamba2 block for MI455X (gfx1250, wave32, WMMA + TDM).
// Pipeline: LN -> in_proj (bf16 WMMA, A staged in LDS by Tensor Data Mover)
//           -> dt/dA prep -> selective scan -> gate+LN
//           -> out_proj (bf16 WMMA) -> clip+LN.
// ---------------------------------------------------------------------------

#define BATCH   4
#define SEQLEN  4096
#define DMODEL  1024
#define DINNER  2048
#define NHEADS  32
#define HEADDIM 64
#define DSTATE  64
#define DPROJ   4256             // 2*DINNER + 2*DSTATE + NHEADS
#define MTOT    (BATCH * SEQLEN) // 16384

typedef __attribute__((ext_vector_type(16))) __bf16 v16bf;
typedef __attribute__((ext_vector_type(8)))  __bf16 v8bf;
typedef __attribute__((ext_vector_type(8)))  float  v8f;
typedef __attribute__((ext_vector_type(4)))  unsigned int u32x4;
typedef __attribute__((ext_vector_type(8)))  int i32x8;
typedef __attribute__((ext_vector_type(4)))  int i32x4;

static __device__ __forceinline__ __bf16 f2bf(float f) {
  unsigned u = __builtin_bit_cast(unsigned, f);
  unsigned r = u + 0x7FFFu + ((u >> 16) & 1u);   // round-to-nearest-even
  unsigned short s = (unsigned short)(r >> 16);
  return __builtin_bit_cast(__bf16, s);
}

static __device__ __forceinline__ float bf2f(__bf16 h) {
  unsigned short s = __builtin_bit_cast(unsigned short, h);
  unsigned u = ((unsigned)s) << 16;
  return __builtin_bit_cast(float, u);
}

// Block-wide sum for 256-thread (8-wave) blocks on wave32.
static __device__ __forceinline__ float block_sum(float v, float* sred) {
  #pragma unroll
  for (int off = 16; off; off >>= 1) v += __shfl_xor(v, off, 32);
  if ((threadIdx.x & 31) == 0) sred[threadIdx.x >> 5] = v;
  __syncthreads();
  float tot = 0.f;
  #pragma unroll
  for (int i = 0; i < 8; ++i) tot += sred[i];
  __syncthreads();
  return tot;
}

// ---------------------------------------------------------------------------
// fp32 -> bf16 weight cast (grid-stride)
// ---------------------------------------------------------------------------
__global__ void cast_bf16_kernel(const float* __restrict__ in,
                                 __bf16* __restrict__ out, int n) {
  for (int i = blockIdx.x * blockDim.x + threadIdx.x; i < n;
       i += gridDim.x * blockDim.x)
    out[i] = f2bf(in[i]);
}

// ---------------------------------------------------------------------------
// Input layernorm over DMODEL, emit bf16 activations for the first GEMM.
// ---------------------------------------------------------------------------
__global__ __launch_bounds__(256) void ln_in_kernel(
    const float* __restrict__ x, const float* __restrict__ w,
    const float* __restrict__ b, __bf16* __restrict__ out) {
  __shared__ float sred[8];
  const size_t row = blockIdx.x;
  const float* xr = x + row * DMODEL;
  float v[4]; float s = 0.f;
  #pragma unroll
  for (int i = 0; i < 4; ++i) { v[i] = xr[threadIdx.x + 256 * i]; s += v[i]; }
  const float mean = block_sum(s, sred) * (1.f / DMODEL);
  float s2 = 0.f;
  #pragma unroll
  for (int i = 0; i < 4; ++i) { float d = v[i] - mean; s2 += d * d; }
  const float inv = rsqrtf(block_sum(s2, sred) * (1.f / DMODEL) + 1e-5f);
  #pragma unroll
  for (int i = 0; i < 4; ++i) {
    const int c = threadIdx.x + 256 * i;
    out[row * DMODEL + c] = f2bf((v[i] - mean) * inv * w[c] + b[c]);
  }
}

// ---------------------------------------------------------------------------
// Generic C[M,N](f32) = A[M,K](bf16) x W[N,K](bf16)^T via v_wmma_f32_16x16x32_bf16.
// The block's shared 16xK A strip is staged into LDS once via the Tensor Data
// Mover (D# built per CDNA5 ISA), WMMA A-fragments are
// then fed from ds_load_b128. One wave -> one 16x64 strip (4 accumulators).
// grid = (ceil(ceil(N/64)/8), M/16), block = 256 (8 waves).
// ---------------------------------------------------------------------------
__global__ __launch_bounds__(256) void gemm_bf16_wmma(
    const __bf16* __restrict__ A, const __bf16* __restrict__ W,
    float* __restrict__ C, int K, int N) {
  __shared__ __align__(16) __bf16 Alds[16 * 2048];   // 64KB; first 16*K used
  const int lane = threadIdx.x & 31;
  const int wave = threadIdx.x >> 5;
  const int m0 = blockIdx.y * 16;
  const __bf16* gsrc = A + (size_t)m0 * K;           // 16 full rows, contiguous

#if __has_builtin(__builtin_amdgcn_tensor_load_to_lds)
  if (wave == 0) {   // one DMA per block; TENSORcnt is per-wave
    const unsigned long long ga = (unsigned long long)(uintptr_t)gsrc;
    const unsigned lds_off = (unsigned)(uintptr_t)Alds;   // low 32 bits = LDS offset
    u32x4 g0 = {
      1u,                                             // count=1 (valid user D#)
      lds_off,                                        // lds_addr
      (unsigned)(ga & 0xFFFFFFFFu),                   // global_addr[31:0]
      (unsigned)((ga >> 32) & 0x1FFFFFFu) | (2u << 30) // global_addr[56:32], type=2
    };
    const unsigned k16 = (unsigned)K & 0xFFFFu;       // K <= 2048 fits 16 bits
    i32x8 g1 = {
      (int)(1u << 16),                 // workgroup_mask=0, data_size=1 (2B)
      (int)(k16 << 16),                // atomic_barrier_addr=0 | tensor_dim0 lo16
      (int)((16384u & 0xFFFFu) << 16), // tensor_dim0 hi16=0 | tensor_dim1 lo16 (MTOT)
      (int)(k16 << 16),                // tensor_dim1 hi16=0 | tile_dim0 = K
      (int)16,                         // tile_dim1 = 16 rows | tile_dim2 = 0
      (int)(unsigned)K,                // tensor_dim0_stride[31:0] = K elements
      0,                               // dim0_stride[47:32]=0 | dim1_stride lo16=0
      0                                // dim1_stride hi
    };
    i32x4 g2 = {0, 0, 0, 0};
    i32x4 g3 = {0, 0, 0, 0};
#if defined(__clang_major__) && (__clang_major__ >= 23)
    i32x8 g4 = {0, 0, 0, 0, 0, 0, 0, 0};
    __builtin_amdgcn_tensor_load_to_lds(g0, g1, g2, g3, g4, 0);
#else
    __builtin_amdgcn_tensor_load_to_lds(g0, g1, g2, g3, 0);
#endif
    __builtin_amdgcn_s_wait_tensorcnt(0);
  }
#else
  {   // fallback: cooperative 16B copies
    const int nvec = 16 * K / 8;
    for (int i = threadIdx.x; i < nvec; i += 256)
      reinterpret_cast<v8bf*>(Alds)[i] = reinterpret_cast<const v8bf*>(gsrc)[i];
  }
#endif
  __syncthreads();

  const int n0 = (blockIdx.x * 8 + wave) * 64;   // wave-uniform
  if (n0 >= N) return;                            // after barrier; EXEC stays full

  const int koffA = (lane >> 4) * 8;    // A chunks at +0 and +16 (ISA A layout)
  const int koffB = (lane >> 4) * 8;    // B chunks at +0 and +16 (same as A)

  int ncol[4];
  #pragma unroll
  for (int t = 0; t < 4; ++t) {
    int nc = n0 + t * 16 + (lane & 15);
    ncol[t] = (nc < N) ? nc : (N - 1);  // clamp; garbage tiles masked at store
  }

  v8f acc[4] = {};
  const __bf16* afrag = Alds + (size_t)(lane & 15) * K + koffA;   // LDS

  for (int k = 0; k < K; k += 32) {
    // Preload all four B fragments so a single wait covers the whole burst.
    v16bf bfr[4];
    #pragma unroll
    for (int t = 0; t < 4; ++t) {
      const __bf16* bptr = W + (size_t)ncol[t] * K + k + koffB;
      const v8bf blo = *reinterpret_cast<const v8bf*>(bptr);
      const v8bf bhi = *reinterpret_cast<const v8bf*>(bptr + 16);
      bfr[t] = __builtin_shufflevector(blo, bhi,
          0,1,2,3,4,5,6,7,8,9,10,11,12,13,14,15);
      __builtin_prefetch(bptr + 64, 0, 1);
    }
    // A fragment from LDS (ds_load_b128 x2).
    const v8bf alo = *reinterpret_cast<const v8bf*>(afrag + k);
    const v8bf ahi = *reinterpret_cast<const v8bf*>(afrag + k + 16);
    const v16bf a = __builtin_shufflevector(alo, ahi,
        0,1,2,3,4,5,6,7,8,9,10,11,12,13,14,15);
    #pragma unroll
    for (int t = 0; t < 4; ++t)
      acc[t] = __builtin_amdgcn_wmma_f32_16x16x32_bf16(
          false, a, false, bfr[t], (short)0, acc[t], false, false);
  }

  const int crow = m0 + ((lane >> 4) * 8);   // C/D layout: VGPR j -> row j (+8 hi-half)
  #pragma unroll
  for (int t = 0; t < 4; ++t) {
    const int nc = n0 + t * 16 + (lane & 15);
    if (nc < N) {
      #pragma unroll
      for (int j = 0; j < 8; ++j)
        C[(size_t)(crow + j) * N + nc] = acc[t][j];
    }
  }
}

// ---------------------------------------------------------------------------
// dt -> dA precompute: dA = min(exp(clip(softplus(dt+bias),1e-3,0.1) * -exp(A_log)), 0.99)
// ---------------------------------------------------------------------------
__global__ __launch_bounds__(256) void prep_kernel(
    const float* __restrict__ zx, const float* __restrict__ A_log,
    const float* __restrict__ dt_bias, float* __restrict__ dAb) {
  const int idx = blockIdx.x * 256 + threadIdx.x;   // MTOT*NHEADS total
  const int h = idx & (NHEADS - 1);
  const size_t row = (size_t)(idx >> 5);
  float dtv = zx[row * DPROJ + (2 * DINNER + 2 * DSTATE) + h] + dt_bias[h];
  float sp = (dtv > 20.f) ? dtv : log1pf(__expf(dtv));
  sp = fminf(fmaxf(sp, 0.001f), 0.1f);
  const float Ah = -__expf(A_log[h]);
  dAb[idx] = fminf(__expf(sp * Ah), 0.99f);
}

// ---------------------------------------------------------------------------
// Selective scan: one workgroup per (batch, head) = 128 WGs, 256 threads.
// State h[64,64] lives in VGPRs: lane owns p = tid>>2, n in [ (tid&3)*16, +16 ).
// N-reduction across 4 consecutive lanes via shuffles (wave32 safe).
// Fused: +D*x residual and SiLU(z) gating; emits bf16 for the second GEMM.
// ---------------------------------------------------------------------------
__global__ __launch_bounds__(256) void scan_kernel(
    const float* __restrict__ zx, const float* __restrict__ dAb,
    const float* __restrict__ Dparam, __bf16* __restrict__ yg) {
  const int bh = blockIdx.x;
  const int b = bh >> 5;
  const int h = bh & (NHEADS - 1);
  const int tid = threadIdx.x;
  const int p  = tid >> 2;
  const int n0 = (tid & 3) * 16;

  float hs[16];
  #pragma unroll
  for (int i = 0; i < 16; ++i) hs[i] = 0.f;

  const float Dp = Dparam[h];
  const float* rowb = zx  + (size_t)b * SEQLEN * DPROJ;
  const float* dAr  = dAb + (size_t)b * SEQLEN * NHEADS + h;
  __bf16* yrow      = yg  + (size_t)b * SEQLEN * DINNER + h * HEADDIM + p;

  for (int t = 0; t < SEQLEN; ++t) {
    const float* r = rowb + (size_t)t * DPROJ;
    const float xv = r[DINNER + h * HEADDIM + p];
    const float da = dAr[(size_t)t * NHEADS];
    const float* Bp = r + 2 * DINNER + n0;
    const float* Cp = r + 2 * DINNER + DSTATE + n0;
    float ysum = 0.f;
    #pragma unroll
    for (int q = 0; q < 4; ++q) {
      const float4 B4 = *reinterpret_cast<const float4*>(Bp + 4 * q);
      const float4 C4 = *reinterpret_cast<const float4*>(Cp + 4 * q);
      float hn;
      hn = fminf(fmaxf(da * hs[4*q+0] + xv * B4.x, -10.f), 10.f); hs[4*q+0] = hn; ysum += C4.x * hn;
      hn = fminf(fmaxf(da * hs[4*q+1] + xv * B4.y, -10.f), 10.f); hs[4*q+1] = hn; ysum += C4.y * hn;
      hn = fminf(fmaxf(da * hs[4*q+2] + xv * B4.z, -10.f), 10.f); hs[4*q+2] = hn; ysum += C4.z * hn;
      hn = fminf(fmaxf(da * hs[4*q+3] + xv * B4.w, -10.f), 10.f); hs[4*q+3] = hn; ysum += C4.w * hn;
    }
    ysum += __shfl_xor(ysum, 1, 32);
    ysum += __shfl_xor(ysum, 2, 32);
    if ((tid & 3) == 0) {
      const float z = r[h * HEADDIM + p];
      const float sil = z / (1.f + __expf(-z));
      yrow[(size_t)t * DINNER] = f2bf((ysum + xv * Dp) * sil);
    }
  }
}

// ---------------------------------------------------------------------------
// Mid layernorm over DINNER, in place on bf16 gated activations.
// ---------------------------------------------------------------------------
__global__ __launch_bounds__(256) void ln_mid_kernel(
    __bf16* __restrict__ y, const float* __restrict__ w,
    const float* __restrict__ b) {
  __shared__ float sred[8];
  const size_t row = blockIdx.x;
  __bf16* yr = y + row * DINNER;
  float v[8]; float s = 0.f;
  #pragma unroll
  for (int i = 0; i < 8; ++i) { v[i] = bf2f(yr[threadIdx.x + 256 * i]); s += v[i]; }
  const float mean = block_sum(s, sred) * (1.f / DINNER);
  float s2 = 0.f;
  #pragma unroll
  for (int i = 0; i < 8; ++i) { float d = v[i] - mean; s2 += d * d; }
  const float inv = rsqrtf(block_sum(s2, sred) * (1.f / DINNER) + 1e-5f);
  #pragma unroll
  for (int i = 0; i < 8; ++i) {
    const int c = threadIdx.x + 256 * i;
    yr[c] = f2bf((v[i] - mean) * inv * w[c] + b[c]);
  }
}

// ---------------------------------------------------------------------------
// Final clip(+-10) + layernorm over DMODEL -> f32 output.
// ---------------------------------------------------------------------------
__global__ __launch_bounds__(256) void ln_out_kernel(
    const float* __restrict__ yin, const float* __restrict__ w,
    const float* __restrict__ b, float* __restrict__ out) {
  __shared__ float sred[8];
  const size_t row = blockIdx.x;
  const float* yr = yin + row * DMODEL;
  float v[4]; float s = 0.f;
  #pragma unroll
  for (int i = 0; i < 4; ++i) {
    v[i] = fminf(fmaxf(yr[threadIdx.x + 256 * i], -10.f), 10.f);
    s += v[i];
  }
  const float mean = block_sum(s, sred) * (1.f / DMODEL);
  float s2 = 0.f;
  #pragma unroll
  for (int i = 0; i < 4; ++i) { float d = v[i] - mean; s2 += d * d; }
  const float inv = rsqrtf(block_sum(s2, sred) * (1.f / DMODEL) + 1e-5f);
  #pragma unroll
  for (int i = 0; i < 4; ++i) {
    const int c = threadIdx.x + 256 * i;
    out[row * DMODEL + c] = (v[i] - mean) * inv * w[c] + b[c];
  }
}

// ---------------------------------------------------------------------------
extern "C" void kernel_launch(void* const* d_in, const int* in_sizes, int n_in,
                              void* d_out, int out_size, void* d_ws, size_t ws_size,
                              hipStream_t stream) {
  (void)in_sizes; (void)n_in; (void)out_size; (void)ws_size;
  const float* x          = (const float*)d_in[0];
  const float* in_proj_w  = (const float*)d_in[1];
  const float* out_proj_w = (const float*)d_in[2];
  const float* A_log      = (const float*)d_in[3];
  const float* dt_bias    = (const float*)d_in[4];
  const float* Dparam     = (const float*)d_in[5];
  const float* ln_w       = (const float*)d_in[6];
  const float* ln_b       = (const float*)d_in[7];
  const float* ni_w       = (const float*)d_in[8];
  const float* ni_b       = (const float*)d_in[9];
  const float* no_w       = (const float*)d_in[10];
  const float* no_b       = (const float*)d_in[11];
  float* out = (float*)d_out;

  char* ws = (char*)d_ws;
  // Workspace carve (all chunks are multiples of 256 B):
  size_t o_xn = 0;                                        // bf16 [MTOT, DMODEL]
  size_t o_w1 = o_xn + (size_t)MTOT  * DMODEL * 2;        // bf16 [DPROJ, DMODEL]
  size_t o_w2 = o_w1 + (size_t)DPROJ * DMODEL * 2;        // bf16 [DMODEL, DINNER]
  size_t o_zx = o_w2 + (size_t)DMODEL * DINNER * 2;       // f32  [MTOT, DPROJ]
  size_t o_dA = o_zx + (size_t)MTOT * DPROJ * 4;          // f32  [MTOT, NHEADS]
  size_t o_yg = o_dA + (size_t)MTOT * NHEADS * 4;         // bf16 [MTOT, DINNER]
  size_t o_op = o_yg + (size_t)MTOT * DINNER * 2;         // f32  [MTOT, DMODEL]

  __bf16* xn  = (__bf16*)(ws + o_xn);
  __bf16* w1  = (__bf16*)(ws + o_w1);
  __bf16* w2  = (__bf16*)(ws + o_w2);
  float*  zxb = (float*) (ws + o_zx);
  float*  dAb = (float*) (ws + o_dA);
  __bf16* yg  = (__bf16*)(ws + o_yg);
  float*  op  = (float*) (ws + o_op);

  // 1) weight casts to bf16
  cast_bf16_kernel<<<4096, 256, 0, stream>>>(in_proj_w,  w1, DPROJ * DMODEL);
  cast_bf16_kernel<<<2048, 256, 0, stream>>>(out_proj_w, w2, DMODEL * DINNER);

  // 2) input layernorm + bf16 cast
  ln_in_kernel<<<MTOT, 256, 0, stream>>>(x, ni_w, ni_b, xn);

  // 3) in_proj GEMM: [MTOT,DPROJ] = xn x w1^T  (ceil(4256/64)=67 waves -> 9 blocks/x)
  gemm_bf16_wmma<<<dim3(9, MTOT / 16), 256, 0, stream>>>(xn, w1, zxb, DMODEL, DPROJ);

  // 4) dt/dA prep
  prep_kernel<<<(MTOT * NHEADS) / 256, 256, 0, stream>>>(zxb, A_log, dt_bias, dAb);

  // 5) selective scan + D residual + SiLU gating (128 independent chains)
  scan_kernel<<<BATCH * NHEADS, 256, 0, stream>>>(zxb, dAb, Dparam, yg);

  // 6) mid layernorm (in place, bf16)
  ln_mid_kernel<<<MTOT, 256, 0, stream>>>(yg, ln_w, ln_b);

  // 7) out_proj GEMM: [MTOT,DMODEL] = yg x w2^T  (1024/64=16 waves -> 2 blocks/x)
  gemm_bf16_wmma<<<dim3(2, MTOT / 16), 256, 0, stream>>>(yg, w2, op, DINNER, DMODEL);

  // 8) clip + final layernorm -> f32 output
  ln_out_kernel<<<MTOT, 256, 0, stream>>>(op, no_w, no_b, out);
}
